// VanillaRNN_10436770529703
// MI455X (gfx1250) — matrix operand order repaired and checked by reading, not compiled, not run on hardware
//
#include <hip/hip_runtime.h>
#include <hip/hip_bf16.h>

typedef __attribute__((ext_vector_type(8)))  _Float16 v8h;
typedef __attribute__((ext_vector_type(16))) _Float16 v16h;
typedef __attribute__((ext_vector_type(8)))  float    v8f;
typedef __attribute__((ext_vector_type(4)))  int      v4i;

#define RNN_B    256
#define RNN_T    512
#define RNN_DIN  64
#define RNN_H    1024
#define RNN_DOUT 10
#define KTILES   34   // (64 + 1024) / 32
#define NTILES   64   // 1024 / 16

// Async global->LDS path (gfx1250) if the toolchain exposes it.
#if defined(__HIP_DEVICE_COMPILE__) && __has_builtin(__builtin_amdgcn_global_load_async_to_lds_b128)
#define USE_ASYNC_LDS 1
#else
#define USE_ASYNC_LDS 0
#endif

__device__ __forceinline__ v16h cat8(v8h lo, v8h hi) {
    return __builtin_shufflevector(lo, hi, 0,1,2,3,4,5,6,7,8,9,10,11,12,13,14,15);
}

#if USE_ASYNC_LDS
// Explicit address-space casts (same pattern HIP's unsafeAtomicAdd uses for
// __builtin_amdgcn_global_atomic_fadd_f64): C-style cast generic -> AS1/AS3,
// with exact pointee type int4 to match the builtin's parameter types.
__device__ __forceinline__ __attribute__((address_space(1))) v4i*
as_g4(const void* p) {
    return (__attribute__((address_space(1))) v4i*)p;
}
__device__ __forceinline__ __attribute__((address_space(3))) v4i*
as_l4(void* p) {
    return (__attribute__((address_space(3))) v4i*)p;
}
__device__ __forceinline__ void wait_async0() {
#if __has_builtin(__builtin_amdgcn_s_wait_asynccnt)
    __builtin_amdgcn_s_wait_asynccnt(0);
#else
    asm volatile("s_wait_asynccnt 0x0" ::: "memory");
#endif
}
#endif

// ---------------------------------------------------------------------------
// Prep 1: pack combined weight matrix W[k][j] (k<64: Whx^T, else Whh^T) into
// per-lane WMMA B-fragment layout: [kt][jt][lane][16 halfs].
// ---------------------------------------------------------------------------
__global__ __launch_bounds__(256) void pack_weights(const float* __restrict__ Whx,
                                                    const float* __restrict__ Whh,
                                                    _Float16* __restrict__ Wpack) {
    int idx = blockIdx.x * 256 + threadIdx.x;
    if (idx >= KTILES * NTILES * 32 * 16) return;
    int i    = idx & 15;
    int lane = (idx >> 4) & 31;
    int jt   = (idx >> 9) & 63;
    int kt   = idx >> 15;
    int k = kt * 32 + ((i >> 3) << 4) + ((lane >> 4) << 3) + (i & 7);
    int j = jt * 16 + (lane & 15);
    float v = (k < RNN_DIN) ? Whx[j * RNN_DIN + k]
                            : Whh[(size_t)j * RNN_H + (k - RNN_DIN)];
    Wpack[idx] = (_Float16)v;
}

// Prep 2: convert x (f32, [B][T][DIN]) to f16.
__global__ __launch_bounds__(256) void cvt_x(const float* __restrict__ x,
                                             _Float16* __restrict__ xf16, int n) {
    int idx = blockIdx.x * 256 + threadIdx.x;
    if (idx < n) xf16[idx] = (_Float16)x[idx];
}

// Prep 3: zero initial hidden state.
__global__ __launch_bounds__(256) void zero_h(_Float16* __restrict__ h, int n) {
    int idx = blockIdx.x * 256 + threadIdx.x;
    if (idx < n) h[idx] = (_Float16)0.0f;
}

// ---------------------------------------------------------------------------
// One recurrence step: hout = tanh([x_t | hin] * Wpack + bias)
// Grid: 32 blocks x 256 threads (8 waves).
// Block (bm, bn): rows bm*128..+128 (wave -> 16 rows), cols bn*64..+64
// (4 N-tiles SHARED by all 8 waves -> staged in LDS, double-buffered).
// ---------------------------------------------------------------------------
__global__ __launch_bounds__(256) void rnn_step(const _Float16* __restrict__ Wpack,
                                                const _Float16* __restrict__ xf16,
                                                const _Float16* __restrict__ hin,
                                                _Float16* __restrict__ hout,
                                                const float* __restrict__ bias,
                                                int t) {
    __shared__ uint4 lbuf[2][256];                 // 2 x 4KB: 4 B-fragments each
    const _Float16* lbh = (const _Float16*)lbuf;   // half view for ds_loads

    const int tid  = threadIdx.x;
    const int lane = tid & 31;
    const int wave = tid >> 5;
    const int bm   = blockIdx.x >> 4;              // 0..1
    const int bn   = blockIdx.x & 15;              // 0..15
    const int mt   = bm * 8 + wave;                // 0..15 : this wave's M tile

    const int mrow = mt * 16 + (lane & 15);        // A-fragment row
    const int klo  = (lane >> 4) << 3;             // 0/8 : A-fragment K sub-off

    const _Float16* xrow = xf16 + ((size_t)mrow * RNN_T + t) * RNN_DIN + klo;
    const _Float16* hrow = hin + (size_t)mrow * RNN_H + klo;
    const uint4* Wp4 = (const uint4*)Wpack;

    // Per K-tile, this block's 4 B fragments are CONTIGUOUS in Wpack:
    // halfs offset (kt*64 + bn*4)*512, i.e. uint4 index (kt*64 + bn*4)*64.
    // 256 threads x 16B = 4KB.
#define STAGE_IDX(KT) ((size_t)((KT) * NTILES + bn * 4) * 64 + tid)

    v8f acc0 = {}, acc1 = {}, acc2 = {}, acc3 = {};

    // A-fragment register prefetch.
    v8h aLo, aHi;
    {
        const _Float16* ap = xrow;                 // kt = 0
        aLo = *(const v8h*)ap; aHi = *(const v8h*)(ap + 16);
    }

    // Prologue: stage kt=0 into buffer 0.
#if USE_ASYNC_LDS
    __builtin_amdgcn_global_load_async_to_lds_b128(as_g4(Wp4 + STAGE_IDX(0)),
                                                   as_l4(&lbuf[0][tid]), 0, 0);
    wait_async0();
#else
    lbuf[0][tid] = Wp4[STAGE_IDX(0)];
#endif
    __syncthreads();

#pragma unroll 2
    for (int kt = 0; kt < KTILES; ++kt) {
        const int nxt = kt + 1;
        // 1) Kick off next K-tile's B copy into the other LDS buffer.
#if USE_ASYNC_LDS
        if (nxt < KTILES)
            __builtin_amdgcn_global_load_async_to_lds_b128(
                as_g4(Wp4 + STAGE_IDX(nxt)), as_l4(&lbuf[nxt & 1][tid]), 0, 0);
#else
        uint4 staged;
        if (nxt < KTILES) staged = Wp4[STAGE_IDX(nxt)];
#endif
        // 2) Register-prefetch next A fragment.
        v8h naLo, naHi;
        if (nxt < KTILES) {
            const _Float16* ap = (nxt < 2) ? (xrow + nxt * 32)
                                           : (hrow + (size_t)(nxt - 2) * 32);
            naLo = *(const v8h*)ap; naHi = *(const v8h*)(ap + 16);
        }
        // 3) Compute current K-tile from LDS.
        const _Float16* lb = lbh + (size_t)(kt & 1) * 2048 + lane * 16;
        v16h a  = cat8(aLo, aHi);
        v16h b0 = *(const v16h*)(lb);
        v16h b1 = *(const v16h*)(lb + 512);
        v16h b2 = *(const v16h*)(lb + 1024);
        v16h b3 = *(const v16h*)(lb + 1536);
        acc0 = __builtin_amdgcn_wmma_f32_16x16x32_f16(false, a, false, b0, (short)0, acc0, false, false);
        acc1 = __builtin_amdgcn_wmma_f32_16x16x32_f16(false, a, false, b1, (short)0, acc1, false, false);
        acc2 = __builtin_amdgcn_wmma_f32_16x16x32_f16(false, a, false, b2, (short)0, acc2, false, false);
        acc3 = __builtin_amdgcn_wmma_f32_16x16x32_f16(false, a, false, b3, (short)0, acc3, false, false);
        aLo = naLo; aHi = naHi;
        // 4) Make next buffer visible to all waves before next iteration.
#if USE_ASYNC_LDS
        wait_async0();
#else
        if (nxt < KTILES) lbuf[nxt & 1][tid] = staged;
#endif
        __syncthreads();
    }
#undef STAGE_IDX

    // Epilogue: D layout -> lanes 0-15: M=r, N=lane; lanes 16-31: M=r+8, N=lane-16.
    const int colbase = bn * 64 + (lane & 15);
    const int rbase   = mt * 16 + ((lane >> 4) << 3);

#define EPILOG(ACC, AIDX)                                                          \
    {                                                                              \
        const int col = colbase + (AIDX) * 16;                                     \
        const float bv = bias[col];                                                \
        _Pragma("unroll")                                                          \
        for (int r = 0; r < 8; ++r) {                                              \
            float v = tanhf(ACC[r] + bv);                                          \
            hout[(size_t)(rbase + r) * RNN_H + col] = (_Float16)v;                 \
        }                                                                          \
    }
    EPILOG(acc0, 0)
    EPILOG(acc1, 1)
    EPILOG(acc2, 2)
    EPILOG(acc3, 3)
#undef EPILOG
}

// ---------------------------------------------------------------------------
// Head: out[b] = softmax(h[b] @ Why^T + Why_b). One block per batch row.
// ---------------------------------------------------------------------------
__global__ __launch_bounds__(256) void rnn_head(const _Float16* __restrict__ h,
                                                const float* __restrict__ Wy,
                                                const float* __restrict__ by,
                                                float* __restrict__ out) {
    __shared__ float o[RNN_DOUT];
    const int b = blockIdx.x, tid = threadIdx.x;
    if (tid < RNN_DOUT) o[tid] = by[tid];
    __syncthreads();

    float acc[RNN_DOUT];
#pragma unroll
    for (int j = 0; j < RNN_DOUT; ++j) acc[j] = 0.0f;
    for (int k = tid; k < RNN_H; k += 256) {
        float hv = (float)h[(size_t)b * RNN_H + k];
#pragma unroll
        for (int j = 0; j < RNN_DOUT; ++j) acc[j] += hv * Wy[j * RNN_H + k];
    }
#pragma unroll
    for (int j = 0; j < RNN_DOUT; ++j) atomicAdd(&o[j], acc[j]);
    __syncthreads();

    if (tid == 0) {
        float mx = o[0];
        for (int j = 1; j < RNN_DOUT; ++j) mx = fmaxf(mx, o[j]);
        float e[RNN_DOUT], s = 0.0f;
        for (int j = 0; j < RNN_DOUT; ++j) { e[j] = __expf(o[j] - mx); s += e[j]; }
        float inv = 1.0f / s;
        for (int j = 0; j < RNN_DOUT; ++j) out[b * RNN_DOUT + j] = e[j] * inv;
    }
}

// ---------------------------------------------------------------------------
extern "C" void kernel_launch(void* const* d_in, const int* in_sizes, int n_in,
                              void* d_out, int out_size, void* d_ws, size_t ws_size,
                              hipStream_t stream) {
    const float* x     = (const float*)d_in[0];
    const float* Whx_w = (const float*)d_in[1];
    const float* Whh_w = (const float*)d_in[2];
    const float* Whh_b = (const float*)d_in[3];
    const float* Why_w = (const float*)d_in[4];
    const float* Why_b = (const float*)d_in[5];
    float* out = (float*)d_out;

    // Workspace layout (bytes):
    //   Wpack : 2,228,224   xf16 : 16,777,216   hbuf0/1 : 524,288 each
    char* ws = (char*)d_ws;
    _Float16* Wpack = (_Float16*)ws;
    _Float16* xf16  = (_Float16*)(ws + 2228224);
    _Float16* h0    = (_Float16*)(ws + 2228224 + 16777216);
    _Float16* h1    = h0 + RNN_B * RNN_H;

    const int npack = KTILES * NTILES * 32 * 16;                 // 1,114,112
    pack_weights<<<npack / 256, 256, 0, stream>>>(Whx_w, Whh_w, Wpack);

    const int nx = RNN_B * RNN_T * RNN_DIN;                      // 8,388,608
    cvt_x<<<nx / 256, 256, 0, stream>>>(x, xf16, nx);

    const int nh = RNN_B * RNN_H;                                // 262,144
    zero_h<<<nh / 256, 256, 0, stream>>>(h0, nh);

    for (int t = 0; t < RNN_T; ++t) {
        const _Float16* hin  = (t & 1) ? h1 : h0;
        _Float16*       hout = (t & 1) ? h0 : h1;
        rnn_step<<<32, 256, 0, stream>>>(Wpack, xf16, hin, hout, Whh_b, t);
    }
    // T=512 is even -> final state is in h0.
    rnn_head<<<RNN_B, 256, 0, stream>>>(h0, Why_w, Why_b, out);
}
